// ExpandingLinear_41300405518782
// MI455X (gfx1250) — matrix operand order repaired and checked by reading, not compiled, not run on hardware
//
#include <hip/hip_runtime.h>
#include <hip/hip_bf16.h>

// ---------------- problem constants ----------------
#define BB   8192
#define F0   2048
#define E1   256
#define E2   256
#define OO   1024
#define F1   2304     // F0 + E1
#define F2   2560     // F1 + E2
#define NNZ  262144
#define NB   1024

// ---------------- GEMM tiling ----------------
#define TM   256      // block tile M (rows of h)
#define TN   128      // block tile N (rows of W = output cols)
#define KT   32       // K slab (bf16 wmma K)
#define KST  40       // padded LDS K-stride (bf16 elems); 80B rows, 16B aligned
#define NK   (F2 / KT)  // 80

typedef __attribute__((ext_vector_type(16))) __bf16 v16bf;
typedef __attribute__((ext_vector_type(8)))  float  v8f;
typedef __attribute__((ext_vector_type(8)))  __bf16 bf16x8;   // 16-byte vector
typedef __attribute__((ext_vector_type(2)))  __bf16 bf16x2;   // one dword (K pair)

// ---------------- workspace layout (bytes) ----------------
#define H_OFF     0ull
#define H_BYTES   ((size_t)BB * F2 * 2)                 // 41,943,040
#define WF_OFF    (H_OFF + H_BYTES)
#define WF_BYTES  ((size_t)OO * F2 * 4)                 // 10,485,760
#define BIAS_OFF  (WF_OFF + WF_BYTES)
#define BIAS_BYTES ((size_t)OO * 4)
#define WB_OFF    (BIAS_OFF + BIAS_BYTES)
#define WB_BYTES  ((size_t)OO * F2 * 2)

// ============================================================
// 1) Build h[B,F2] in bf16: [x | relu(x[e1p]*e1w) | relu(h[e2p]*e2w)]
// ============================================================
__global__ __launch_bounds__(256) void build_h_kernel(
    const float* __restrict__ x, const float* __restrict__ e1w,
    const float* __restrict__ e2w, const int* __restrict__ e1p,
    const int* __restrict__ e2p, __bf16* __restrict__ h)
{
  int idx = blockIdx.x * 256 + threadIdx.x;      // < B*F2 = 20,971,520
  int col = idx % F2;
  int row = idx / F2;
  const float* xr = x + (size_t)row * F0;
  float v;
  if (col < F0) {
    v = xr[col];
  } else if (col < F1) {
    int j = col - F0;
    v = fmaxf(xr[e1p[j]] * e1w[j], 0.0f);
  } else {
    int j = col - F1;
    int p = e2p[j];
    float pv;
    if (p < F0) {
      pv = xr[p];
    } else {
      int q = p - F0;
      pv = fmaxf(xr[e1p[q]] * e1w[q], 0.0f);
    }
    v = fmaxf(pv * e2w[j], 0.0f);
  }
  h[(size_t)row * F2 + col] = (__bf16)v;
}

// ============================================================
// 2) Densify W (COO -> dense f32) and bias
// ============================================================
__global__ __launch_bounds__(256) void zero_kernel(float* __restrict__ p, int n)
{
  int i = blockIdx.x * 256 + threadIdx.x;
  if (i < n) p[i] = 0.0f;
}

__global__ __launch_bounds__(256) void scatter_w_kernel(
    const int* __restrict__ rows, const int* __restrict__ cols,
    const float* __restrict__ vals, float* __restrict__ Wf)
{
  int i = blockIdx.x * 256 + threadIdx.x;
  if (i < NNZ) atomicAdd(&Wf[(size_t)rows[i] * F2 + cols[i]], vals[i]);
}

__global__ __launch_bounds__(256) void scatter_b_kernel(
    const int* __restrict__ bidx, const float* __restrict__ bvals,
    float* __restrict__ bias)
{
  int i = blockIdx.x * 256 + threadIdx.x;
  if (i < NB) atomicAdd(&bias[bidx[i]], bvals[i]);
}

__global__ __launch_bounds__(256) void convert_w_kernel(
    const float* __restrict__ Wf, __bf16* __restrict__ Wb)
{
  int i = blockIdx.x * 256 + threadIdx.x;   // < O*F2
  Wb[i] = (__bf16)Wf[i];
}

// ============================================================
// 3) GEMM: out[B,O] = h @ Wb^T + bias, v_wmma_f32_16x16x32_bf16
//    Block: 256 threads (8 waves), 256x128 output tile, double-buffered LDS.
//    Waves: 4 along M x 2 along N; wave tile 64x64 = 4x4 WMMA tiles.
//    Pipeline: global loads (next slab) -> 16 WMMAs (current buf) ->
//              LDS stores (other buf) -> one barrier per K iteration.
// ============================================================
__global__ __launch_bounds__(256) void gemm_wmma_kernel(
    const __bf16* __restrict__ H, const __bf16* __restrict__ Wb,
    const float* __restrict__ bias, float* __restrict__ out)
{
  __shared__ __bf16 As[2][TM * KST];   // 2 * 256*40*2 = 40960 B, [m][k]
  __shared__ __bf16 Bs[2][TN * KST];   // 2 * 128*40*2 = 20480 B, [n][k]

  const int tid  = threadIdx.x;
  const int lane = tid & 31;
  const int wave = tid >> 5;          // 0..7
  const int wm   = wave & 3;          // 4 waves along M
  const int wn   = wave >> 2;         // 2 waves along N
  const int half = lane >> 4;
  const int l15  = lane & 15;

  const int bm = blockIdx.y * TM;
  const int bn = blockIdx.x * TN;

  v8f acc[4][4];
  const v8f vzero = {0.f, 0.f, 0.f, 0.f, 0.f, 0.f, 0.f, 0.f};
#pragma unroll
  for (int i = 0; i < 4; ++i)
#pragma unroll
    for (int j = 0; j < 4; ++j) acc[i][j] = vzero;

  // tile-load coords: 4 threads x 16B cover one 64B (32-elem) K row
  const int kc = (tid & 3) * 8;       // 0,8,16,24 (bf16 elems)
  const int r0 = tid >> 2;            // 0..63

  const __bf16* Arow0 = H  + (size_t)(bm + r0) * F2 + kc;   // +64/128/192 rows
  const __bf16* Brow0 = Wb + (size_t)(bn + r0) * F2 + kc;   // +64 rows

  // ---- preload slab 0 into buffer 0 ----
  {
    *(bf16x8*)&As[0][(r0      ) * KST + kc] = *(const bf16x8*)(Arow0 + (size_t)  0 * F2);
    *(bf16x8*)&As[0][(r0 +  64) * KST + kc] = *(const bf16x8*)(Arow0 + (size_t) 64 * F2);
    *(bf16x8*)&As[0][(r0 + 128) * KST + kc] = *(const bf16x8*)(Arow0 + (size_t)128 * F2);
    *(bf16x8*)&As[0][(r0 + 192) * KST + kc] = *(const bf16x8*)(Arow0 + (size_t)192 * F2);
    *(bf16x8*)&Bs[0][(r0      ) * KST + kc] = *(const bf16x8*)(Brow0 + (size_t)  0 * F2);
    *(bf16x8*)&Bs[0][(r0 +  64) * KST + kc] = *(const bf16x8*)(Brow0 + (size_t) 64 * F2);
  }
  __syncthreads();

  for (int kt = 0; kt < NK; ++kt) {
    const int buf = kt & 1;
    const int nxt = buf ^ 1;
    const bool more = (kt + 1 < NK);
    const int koff = (kt + 1) * KT;

    // ---- issue global loads for next slab (overlap with WMMA below) ----
    bf16x8 sa0, sa1, sa2, sa3, sb0, sb1;
    if (more) {
      sa0 = *(const bf16x8*)(Arow0 + (size_t)  0 * F2 + koff);
      sa1 = *(const bf16x8*)(Arow0 + (size_t) 64 * F2 + koff);
      sa2 = *(const bf16x8*)(Arow0 + (size_t)128 * F2 + koff);
      sa3 = *(const bf16x8*)(Arow0 + (size_t)192 * F2 + koff);
      sb0 = *(const bf16x8*)(Brow0 + (size_t)  0 * F2 + koff);
      sb1 = *(const bf16x8*)(Brow0 + (size_t) 64 * F2 + koff);
    }

    // ---- B fragments (held across all i), per ISA 16-bit 32x16 layout ----
    v16bf bfrag[4];
#pragma unroll
    for (int j = 0; j < 4; ++j) {
      const __bf16* brow = &Bs[buf][(wn * 64 + j * 16 + l15) * KST];
#pragma unroll
      for (int v = 0; v < 8; ++v) {
        int kk = (v < 4) ? (2 * v + 8 * half) : (16 + 2 * (v - 4) + 8 * half);
        bf16x2 p = *(const bf16x2*)&brow[kk];
        bfrag[j][2 * v]     = p[0];
        bfrag[j][2 * v + 1] = p[1];
      }
    }

    // ---- A fragment per i, 4 WMMAs each (16 wmma / iter / wave) ----
#pragma unroll
    for (int i = 0; i < 4; ++i) {
      const __bf16* arow = &As[buf][(wm * 64 + i * 16 + l15) * KST];
      v16bf af;
#pragma unroll
      for (int v = 0; v < 8; ++v) {
        int kk = (v < 4) ? (2 * v + 8 * half) : (16 + 2 * (v - 4) + 8 * half);
        bf16x2 p = *(const bf16x2*)&arow[kk];
        af[2 * v]     = p[0];
        af[2 * v + 1] = p[1];
      }
#pragma unroll
      for (int j = 0; j < 4; ++j)
        acc[i][j] = __builtin_amdgcn_wmma_f32_16x16x32_bf16(
            false, af, false, bfrag[j], (short)0, acc[i][j], false, false);
    }

    // ---- commit staged slab into the other buffer ----
    if (more) {
      *(bf16x8*)&As[nxt][(r0      ) * KST + kc] = sa0;
      *(bf16x8*)&As[nxt][(r0 +  64) * KST + kc] = sa1;
      *(bf16x8*)&As[nxt][(r0 + 128) * KST + kc] = sa2;
      *(bf16x8*)&As[nxt][(r0 + 192) * KST + kc] = sa3;
      *(bf16x8*)&Bs[nxt][(r0      ) * KST + kc] = sb0;
      *(bf16x8*)&Bs[nxt][(r0 +  64) * KST + kc] = sb1;
    }
    __syncthreads();
  }

  // ---- epilogue: C/D layout VGPR r -> (M = r + 8*half, N = l15); add bias ----
#pragma unroll
  for (int j = 0; j < 4; ++j) {
    const int gn = bn + wn * 64 + j * 16 + l15;
    const float bv = bias[gn];
#pragma unroll
    for (int i = 0; i < 4; ++i) {
      const int mbase = bm + wm * 64 + i * 16 + half * 8;
#pragma unroll
      for (int r = 0; r < 8; ++r) {
        out[(size_t)(mbase + r) * OO + gn] = acc[i][j][r] + bv;
      }
    }
  }
}

// ============================================================
// launcher
// ============================================================
extern "C" void kernel_launch(void* const* d_in, const int* in_sizes, int n_in,
                              void* d_out, int out_size, void* d_ws, size_t ws_size,
                              hipStream_t stream)
{
  const float* x       = (const float*)d_in[0];
  const float* e1_w    = (const float*)d_in[1];
  const float* e2_w    = (const float*)d_in[2];
  const float* w_vals  = (const float*)d_in[3];
  const float* b_vals  = (const float*)d_in[4];
  const int*   e1_p    = (const int*)  d_in[5];
  const int*   e2_p    = (const int*)  d_in[6];
  const int*   w_rows  = (const int*)  d_in[7];
  const int*   w_cols  = (const int*)  d_in[8];
  const int*   b_idx   = (const int*)  d_in[9];
  float* out = (float*)d_out;

  char* ws = (char*)d_ws;
  __bf16* h_bf   = (__bf16*)(ws + H_OFF);
  float*  Wf     = (float*) (ws + WF_OFF);
  float*  bias   = (float*) (ws + BIAS_OFF);
  __bf16* Wb     = (__bf16*)(ws + WB_OFF);

  // 1) build h (bf16)
  build_h_kernel<<<(BB * F2) / 256, 256, 0, stream>>>(x, e1_w, e2_w, e1_p, e2_p, h_bf);

  // 2) densify W + bias (contiguous f32 region: O*F2 + O)
  {
    int nzero = OO * F2 + OO;
    zero_kernel<<<(nzero + 255) / 256, 256, 0, stream>>>(Wf, nzero);
  }
  scatter_w_kernel<<<(NNZ + 255) / 256, 256, 0, stream>>>(w_rows, w_cols, w_vals, Wf);
  scatter_b_kernel<<<(NB + 255) / 256, 256, 0, stream>>>(b_idx, b_vals, bias);
  convert_w_kernel<<<(OO * F2) / 256, 256, 0, stream>>>(Wf, Wb);

  // 3) WMMA GEMM: grid (O/TN, B/TM) = (8, 32)
  dim3 grid(OO / TN, BB / TM);
  gemm_wmma_kernel<<<grid, 256, 0, stream>>>(h_bf, Wb, bias, out);
}
